// Contextual_Loss_14233521619451
// MI455X (gfx1250) — matrix operand order repaired and checked by reading, not compiled, not run on hardware
//
#include <hip/hip_runtime.h>
#include <math.h>

typedef __attribute__((ext_vector_type(16))) _Float16 v16h;
typedef __attribute__((ext_vector_type(8)))  float    v8f;

#define N_BATCH 2
#define C_CH    256
#define S_POS   6400       // 80*80
#define NB      400        // S_POS / 16
#define KB      8          // C_CH / 32
#define CH      8          // chunks per strip (parallelism split)
#define JT      (NB / CH)  // 50 tiles per wave
#define NS      (N_BATCH * S_POS)   // 12800
#define EPSF    1e-5f

// workspace layout (byte offsets)
#define OFF_MEANT    0u
#define OFF_ROWSCALE 4096u                    // f32[NS]  : 10/(rowmin+eps)
#define OFF_INVZ     57344u                   // f32[NS]  : 1/Z
#define OFF_COLMAX   110592u                  // f32[NS]
#define OFF_PMIN     163840u                  // f32[CH][NS] partials
#define OFF_PSUM     573440u
#define OFF_PMAX     983040u
#define OFF_ASWZ     1572864u                 // swizzled f16 A fragments
#define ASWZ_BYTES   ((unsigned)(N_BATCH*NB*KB)*1024u)   // 6,553,600
#define OFF_BSWZ     (OFF_ASWZ + ASWZ_BYTES)

// ---------------- mean over (N, H, W) per channel ----------------
__global__ void k_meanT(const float* __restrict__ T, float* __restrict__ meanT) {
  __shared__ float red[256];
  int c = blockIdx.x, t = threadIdx.x;
  float s = 0.f;
  for (int i = t; i < N_BATCH * S_POS; i += 256) {
    int n = i / S_POS, sp = i - n * S_POS;
    s += T[((size_t)(n * C_CH + c)) * S_POS + sp];
  }
  red[t] = s; __syncthreads();
  for (int o = 128; o > 0; o >>= 1) { if (t < o) red[t] += red[t + o]; __syncthreads(); }
  if (t == 0) meanT[c] = red[0] / (float)(N_BATCH * S_POS);
}

// ------- center, L2-normalize over C, emit WMMA-swizzled f16 fragments -------
// A fragment layout (16-bit A 16x32): lane L: M=L&15, half=L>>4,
//   dword v<4 : K = 8*half + 2*v + {0,1};  v>=4 : K = 16 + 8*half + 2*(v-4) + {0,1}
// B fragment layout (16-bit B 32x16): lane L: N=L&15, half=L>>4,
//   dword v<4 : K = 8*half + 2*v + {0,1};  v>=4 : K = 16 + 8*half + 2*(v-4) + {0,1}
__global__ void k_normswz(const float* __restrict__ I, const float* __restrict__ T,
                          const float* __restrict__ meanT,
                          unsigned* __restrict__ Aswz, unsigned* __restrict__ Bswz) {
  __shared__ float sI[C_CH][16];
  __shared__ float sT[C_CH][16];
  __shared__ float invI[16], invT[16];
  int b = blockIdx.x;
  int n = b / NB, ib = b - n * NB;
  int t = threadIdx.x;
  int m = t & 15, cg = t >> 4;
  for (int it = 0; it < 16; ++it) {
    int c = it * 16 + cg;
    float mu = meanT[c];
    size_t base = ((size_t)(n * C_CH + c)) * S_POS + ib * 16 + m;
    sI[c][m] = I[base] - mu;
    sT[c][m] = T[base] - mu;
  }
  __syncthreads();
  if (t < 32) {
    int mm = t & 15;
    float s = 0.f;
    if (t < 16) { for (int c = 0; c < C_CH; ++c) { float v = sI[c][mm]; s += v * v; } invI[mm] = 1.0f / sqrtf(s); }
    else        { for (int c = 0; c < C_CH; ++c) { float v = sT[c][mm]; s += v * v; } invT[mm] = 1.0f / sqrtf(s); }
  }
  __syncthreads();
  unsigned fb = (unsigned)(n * NB + ib) * KB;
  for (int u = 0; u < 8; ++u) {
    int q = t + 256 * u;                 // 0..2047
    int kb = q >> 8, rem = q & 255;
    int L = rem >> 3, v = rem & 7;
    int half = L >> 4, ll = L & 15;
    union { _Float16 h[2]; unsigned u32; } pa, pb;
    int kA = (v < 4) ? (8 * half + 2 * v) : (16 + 8 * half + 2 * (v - 4));
    int cA = kb * 32 + kA;
    pa.h[0] = (_Float16)(sI[cA][ll]     * invI[ll]);
    pa.h[1] = (_Float16)(sI[cA + 1][ll] * invI[ll]);
    Aswz[(size_t)(fb + kb) * 256u + rem] = pa.u32;
    int kB = (v < 4) ? (8 * half + 2 * v) : (16 + 8 * half + 2 * (v - 4));
    int cB = kb * 32 + kB;
    pb.h[0] = (_Float16)(sT[cB][ll]     * invT[ll]);
    pb.h[1] = (_Float16)(sT[cB + 1][ll] * invT[ll]);
    Bswz[(size_t)(fb + kb) * 256u + rem] = pb.u32;
  }
}

__device__ __forceinline__ const v16h* fragPtr(const char* base, unsigned f, int lane) {
  return (const v16h*)(base + (size_t)f * 1024u + (unsigned)lane * 32u);
}
__device__ __forceinline__ v8f tile_wmma(const v16h a[KB], const char* Bb, unsigned bfb, int lane) {
  v16h bf[KB];
#pragma unroll
  for (int kb = 0; kb < KB; ++kb) bf[kb] = *fragPtr(Bb, bfb + kb, lane);
  v8f acc = {};
#pragma unroll
  for (int kb = 0; kb < KB; ++kb)
    acc = __builtin_amdgcn_wmma_f32_16x16x32_f16(false, a[kb], false, bf[kb],
                                                 (short)0, acc, false, false);
  return acc;
}

// pass kernels: block b -> (n, ch, group-of-8); wave-in-block selects strip.
// All 8 waves of a block share n and the streamed fragment sequence -> WGP$ reuse.

// ---------------- pass 1: partial row-wise min of raw distance ----------------
__global__ void k_rowmin(const char* __restrict__ Ab, const char* __restrict__ Bb,
                         float* __restrict__ Pmin) {
  int b = blockIdx.x;                          // N_BATCH*CH*(NB/8) = 800
  int w = threadIdx.x >> 5, lane = threadIdx.x & 31;
  int n = b / (CH * (NB / 8)); int rem = b - n * (CH * (NB / 8));
  int ch = rem / (NB / 8);      int ibg = rem - ch * (NB / 8);
  int ib = ibg * 8 + w;
  unsigned afb = (unsigned)(n * NB + ib) * KB;
  v16h a[KB];
#pragma unroll
  for (int kb = 0; kb < KB; ++kb) a[kb] = *fragPtr(Ab, afb + kb, lane);
  float rmin[8];
#pragma unroll
  for (int r = 0; r < 8; ++r) rmin[r] = 3.4e38f;
  for (int jt = 0; jt < JT; ++jt) {
    int jb = ch * JT + jt;
    v8f acc = tile_wmma(a, Bb, (unsigned)(n * NB + jb) * KB, lane);
#pragma unroll
    for (int r = 0; r < 8; ++r) {
      float raw = fmaxf(0.f, 0.5f * (1.f - acc[r]));
      rmin[r] = fminf(rmin[r], raw);
    }
  }
#pragma unroll
  for (int mk = 1; mk <= 8; mk <<= 1)
#pragma unroll
    for (int r = 0; r < 8; ++r) rmin[r] = fminf(rmin[r], __shfl_xor(rmin[r], mk, 32));
  if ((lane & 15) == 0) {
    int rbase = ch * NS + n * S_POS + ib * 16 + 8 * (lane >> 4);
#pragma unroll
    for (int r = 0; r < 8; ++r) Pmin[rbase + r] = rmin[r];
  }
}

// ---------------- pass 2: partial row-wise sum of exp weights ----------------
__global__ void k_rowsum(const char* __restrict__ Ab, const char* __restrict__ Bb,
                         const float* __restrict__ rowscale, float* __restrict__ Psum) {
  int b = blockIdx.x;
  int w = threadIdx.x >> 5, lane = threadIdx.x & 31;
  int n = b / (CH * (NB / 8)); int rem = b - n * (CH * (NB / 8));
  int ch = rem / (NB / 8);      int ibg = rem - ch * (NB / 8);
  int ib = ibg * 8 + w;
  int half = lane >> 4;
  unsigned afb = (unsigned)(n * NB + ib) * KB;
  v16h a[KB];
#pragma unroll
  for (int kb = 0; kb < KB; ++kb) a[kb] = *fragPtr(Ab, afb + kb, lane);
  float sc[8], rsum[8];
#pragma unroll
  for (int r = 0; r < 8; ++r) {
    sc[r] = rowscale[n * S_POS + ib * 16 + 8 * half + r];
    rsum[r] = 0.f;
  }
  for (int jt = 0; jt < JT; ++jt) {
    int jb = ch * JT + jt;
    v8f acc = tile_wmma(a, Bb, (unsigned)(n * NB + jb) * KB, lane);
#pragma unroll
    for (int r = 0; r < 8; ++r) {
      float raw = fmaxf(0.f, 0.5f * (1.f - acc[r]));
      rsum[r] += __expf(10.0f - raw * sc[r]);
    }
  }
#pragma unroll
  for (int mk = 1; mk <= 8; mk <<= 1)
#pragma unroll
    for (int r = 0; r < 8; ++r) rsum[r] += __shfl_xor(rsum[r], mk, 32);
  if ((lane & 15) == 0) {
    int rbase = ch * NS + n * S_POS + ib * 16 + 8 * half;
#pragma unroll
    for (int r = 0; r < 8; ++r) Psum[rbase + r] = rsum[r];
  }
}

// ---------------- pass 3: partial column-wise max of cs = e * invZ ----------------
__global__ void k_colmax(const char* __restrict__ Ab, const char* __restrict__ Bb,
                         const float* __restrict__ rowscale, const float* __restrict__ invZ,
                         float* __restrict__ Pmax) {
  int b = blockIdx.x;
  int w = threadIdx.x >> 5, lane = threadIdx.x & 31;
  int n = b / (CH * (NB / 8)); int rem = b - n * (CH * (NB / 8));
  int ch = rem / (NB / 8);      int jbg = rem - ch * (NB / 8);
  int jb = jbg * 8 + w;
  int half = lane >> 4;
  unsigned bfb = (unsigned)(n * NB + jb) * KB;
  v16h bfr[KB];
#pragma unroll
  for (int kb = 0; kb < KB; ++kb) bfr[kb] = *fragPtr(Bb, bfb + kb, lane);
  float rmax = 0.f;
  for (int it = 0; it < JT; ++it) {
    int ib = ch * JT + it;
    unsigned afb = (unsigned)(n * NB + ib) * KB;
    v16h a[KB];
#pragma unroll
    for (int kb = 0; kb < KB; ++kb) a[kb] = *fragPtr(Ab, afb + kb, lane);
    v8f acc = {};
#pragma unroll
    for (int kb = 0; kb < KB; ++kb)
      acc = __builtin_amdgcn_wmma_f32_16x16x32_f16(false, a[kb], false, bfr[kb],
                                                   (short)0, acc, false, false);
    int rbase = n * S_POS + ib * 16 + 8 * half;
#pragma unroll
    for (int r = 0; r < 8; ++r) {
      float raw = fmaxf(0.f, 0.5f * (1.f - acc[r]));
      float cs  = __expf(10.0f - raw * rowscale[rbase + r]) * invZ[rbase + r];
      rmax = fmaxf(rmax, cs);
    }
  }
  rmax = fmaxf(rmax, __shfl_xor(rmax, 16, 32));
  if (lane < 16) Pmax[ch * NS + n * S_POS + jb * 16 + lane] = rmax;
}

// ---------------- deterministic partial combines ----------------
__global__ void k_redmin(const float* __restrict__ P, float* __restrict__ rowscale) {
  int i = blockIdx.x * 256 + threadIdx.x;      // NS = 12800 -> 50 blocks
  float m = P[i];
#pragma unroll
  for (int c = 1; c < CH; ++c) m = fminf(m, P[c * NS + i]);
  rowscale[i] = 10.0f / (m + EPSF);
}
__global__ void k_redsum(const float* __restrict__ P, float* __restrict__ invZ) {
  int i = blockIdx.x * 256 + threadIdx.x;
  float s = 0.f;
#pragma unroll
  for (int c = 0; c < CH; ++c) s += P[c * NS + i];
  invZ[i] = 1.0f / s;
}
__global__ void k_redmax(const float* __restrict__ P, float* __restrict__ colmax) {
  int i = blockIdx.x * 256 + threadIdx.x;
  float m = P[i];
#pragma unroll
  for (int c = 1; c < CH; ++c) m = fmaxf(m, P[c * NS + i]);
  colmax[i] = m;
}

// ---------------- final: loss = mean_n(-log(mean_j colmax)) ----------------
__global__ void k_final(const float* __restrict__ colmax, float* __restrict__ out) {
  __shared__ float red[256];
  __shared__ float lossAcc;
  int t = threadIdx.x;
  if (t == 0) lossAcc = 0.f;
  __syncthreads();
  for (int n = 0; n < N_BATCH; ++n) {
    float s = 0.f;
    for (int j = t; j < S_POS; j += 256) s += colmax[n * S_POS + j];
    red[t] = s; __syncthreads();
    for (int o = 128; o > 0; o >>= 1) { if (t < o) red[t] += red[t + o]; __syncthreads(); }
    if (t == 0) lossAcc += -logf(red[0] / (float)S_POS);
    __syncthreads();
  }
  if (t == 0) out[0] = lossAcc / (float)N_BATCH;
}

extern "C" void kernel_launch(void* const* d_in, const int* in_sizes, int n_in,
                              void* d_out, int out_size, void* d_ws, size_t ws_size,
                              hipStream_t stream) {
  (void)in_sizes; (void)n_in; (void)out_size; (void)ws_size;
  const float* I = (const float*)d_in[0];
  const float* T = (const float*)d_in[1];
  char* ws = (char*)d_ws;
  float*    meanT    = (float*)(ws + OFF_MEANT);
  float*    rowscale = (float*)(ws + OFF_ROWSCALE);
  float*    invZ     = (float*)(ws + OFF_INVZ);
  float*    colmax   = (float*)(ws + OFF_COLMAX);
  float*    Pmin     = (float*)(ws + OFF_PMIN);
  float*    Psum     = (float*)(ws + OFF_PSUM);
  float*    Pmax     = (float*)(ws + OFF_PMAX);
  unsigned* Aswz     = (unsigned*)(ws + OFF_ASWZ);
  unsigned* Bswz     = (unsigned*)(ws + OFF_BSWZ);
  const char* Ab = (const char*)Aswz;
  const char* Bb = (const char*)Bswz;

  k_meanT  <<<C_CH,         256, 0, stream>>>(T, meanT);
  k_normswz<<<N_BATCH * NB, 256, 0, stream>>>(I, T, meanT, Aswz, Bswz);

  int gemmBlocks = N_BATCH * CH * (NB / 8);   // 800 blocks x 8 waves = 6400 waves
  int redBlocks  = NS / 256;                  // 50

  k_rowmin<<<gemmBlocks, 256, 0, stream>>>(Ab, Bb, Pmin);
  k_redmin<<<redBlocks,  256, 0, stream>>>(Pmin, rowscale);
  k_rowsum<<<gemmBlocks, 256, 0, stream>>>(Ab, Bb, rowscale, Psum);
  k_redsum<<<redBlocks,  256, 0, stream>>>(Psum, invZ);
  k_colmax<<<gemmBlocks, 256, 0, stream>>>(Ab, Bb, rowscale, invZ, Pmax);
  k_redmax<<<redBlocks,  256, 0, stream>>>(Pmax, colmax);
  k_final <<<1,          256, 0, stream>>>(colmax, (float*)d_out);
}
